// KPRN_24472723653139
// MI455X (gfx1250) — compile-verified, matrix operand order checked by reading, not run on hardware
//
#include <hip/hip_runtime.h>

typedef _Float16 half8 __attribute__((ext_vector_type(8)));
typedef _Float16 v16h  __attribute__((ext_vector_type(16)));
typedef float    v8f   __attribute__((ext_vector_type(8)));

#define BB 128
#define CC 5
#define PP 5
#define LL 5
#define DD 128
#define NN (BB*CC)     // 640 sequences
#define F0 384         // layer0 input features
#define K0 512         // layer0 fused K (x 384 + h 128)
#define K1 256         // layer1 fused K (h0 128 + h1 128)
#define G4 512         // 4*D gate width

__device__ __forceinline__ float sigf(float x) { return 1.0f / (1.0f + __expf(-x)); }

// ---------------------------------------------------------------------------
// Kernel 1: gather node/type/relation embeddings into f16 X[p][n][l][384]
// ---------------------------------------------------------------------------
__global__ void build_x_kernel(const int* __restrict__ user_index,
                               const int* __restrict__ paths,
                               const int* __restrict__ types,
                               const int* __restrict__ rels,
                               const float* __restrict__ entity_emb,
                               const float* __restrict__ relation_emb,
                               const float* __restrict__ user_emb,
                               const float* __restrict__ news_emb,
                               const float* __restrict__ type_emb,
                               _Float16* __restrict__ Xg)
{
    int t = blockIdx.x;            // flat (b,c,p,l), 16000 blocks
    int l = t % LL; t /= LL;
    int p = t % PP; t /= PP;
    int c = t % CC; t /= CC;
    int b = t;
    int f = threadIdx.x;           // 0..127

    int uid = user_index[b];
    long base = (((long)uid * CC + c) * PP + p) * LL + l;
    int path = paths[base];
    int ty   = types[base];
    int rl   = rels[base];

    const float* node = (ty == 0) ? (user_emb   + (long)path * DD)
                      : (ty == 1) ? (news_emb   + (long)path * DD)
                                  : (entity_emb + (long)path * DD);

    int n = b * CC + c;
    long o = (((long)p * NN + n) * LL + l) * F0;
    Xg[o + f]         = (_Float16)node[f];
    Xg[o + 128 + f]   = (_Float16)type_emb[ty * DD + f];
    Xg[o + 256 + f]   = (_Float16)relation_emb[rl * DD + f];
}

// ---------------------------------------------------------------------------
// Kernel 2: concat + convert weights to f16 (row-major [4D][K], K = in||hid),
// fuse biases.
// ---------------------------------------------------------------------------
__global__ void build_w_kernel(const float* __restrict__ Wih0, const float* __restrict__ Whh0,
                               const float* __restrict__ bih0, const float* __restrict__ bhh0,
                               const float* __restrict__ Wih1, const float* __restrict__ Whh1,
                               const float* __restrict__ bih1, const float* __restrict__ bhh1,
                               _Float16* __restrict__ Wc0, _Float16* __restrict__ Wc1,
                               float* __restrict__ b0, float* __restrict__ b1)
{
    int pj = blockIdx.x;           // p*512 + gate-row j, 2560 blocks
    long r0 = (long)pj * K0;
    long r1 = (long)pj * K1;
    for (int k = threadIdx.x; k < K0; k += blockDim.x) {
        float v = (k < F0) ? Wih0[(long)pj * F0 + k] : Whh0[(long)pj * DD + (k - F0)];
        Wc0[r0 + k] = (_Float16)v;
    }
    for (int k = threadIdx.x; k < K1; k += blockDim.x) {
        float v = (k < DD) ? Wih1[(long)pj * DD + k] : Whh1[(long)pj * DD + (k - DD)];
        Wc1[r1 + k] = (_Float16)v;
    }
    if (threadIdx.x == 0) {
        b0[pj] = bih0[pj] + bhh0[pj];
        b1[pj] = bih1[pj] + bhh1[pj];
    }
}

// ---------------------------------------------------------------------------
// Kernel 3: 2-layer LSTM via v_wmma_f32_16x16x32_f16.
// Block = 16 sequence rows of one path p. 8 waves; wave w owns hidden units
// [16w,16w+16) for all four gates -> cell update fully in registers.
// Weight fragments are re-fetched each timestep with a t-dependent k-tile
// rotation so LICM cannot hoist them out of the t loop (avoids scratch spill;
// the fetches hit L2 since all 40 row-blocks share the same 4MB of weights).
// ---------------------------------------------------------------------------
__launch_bounds__(256)
__global__ void lstm_kernel(const _Float16* __restrict__ Xg,
                            const _Float16* __restrict__ Wc0,
                            const _Float16* __restrict__ Wc1,
                            const float* __restrict__ b0,
                            const float* __restrict__ b1,
                            float* __restrict__ sel)
{
    __shared__ _Float16 A0[16][K0];   // [x_t (384) || h0_prev (128)] per row, 16KB
    __shared__ _Float16 A1[16][K1];   // [h0_t (128) || h1_prev (128)] per row, 8KB

    const int tid  = threadIdx.x;
    const int wave = tid >> 5;
    const int lane = tid & 31;
    const int p    = blockIdx.y;           // path index
    const int n0   = blockIdx.x * 16;      // first sequence row of this block

    const int u    = (wave << 4) + (lane & 15);  // hidden unit 0..127
    const int hiH  = lane >> 4;                  // lane half select
    const int arow = lane & 15;                  // A-fragment row
    const int kb   = hiH ? 8 : 0;                // A/B fragment K base

    const _Float16* W0p = Wc0 + (long)p * G4 * K0;
    const _Float16* W1p = Wc1 + (long)p * G4 * K1;
    const float* b0p = b0 + p * G4;
    const float* b1p = b1 + p * G4;

    const float bi0 = b0p[u], bf0 = b0p[128 + u], bg0 = b0p[256 + u], bo0 = b0p[384 + u];
    const float bi1 = b1p[u], bf1 = b1p[128 + u], bg1 = b1p[256 + u], bo1 = b1p[384 + u];

    // zero the hidden-state portions of the LDS activation buffers
    for (int i = tid; i < 16 * 128; i += 256) A0[i >> 7][F0 + (i & 127)] = (_Float16)0.0f;
    for (int i = tid; i < 16 * K1;  i += 256) A1[i >> 8][i & 255]        = (_Float16)0.0f;

    float c0s[8], c1s[8];
    #pragma unroll
    for (int v = 0; v < 8; ++v) { c0s[v] = 0.0f; c1s[v] = 0.0f; }

    const v8f vzero = {0.f, 0.f, 0.f, 0.f, 0.f, 0.f, 0.f, 0.f};

    for (int t = 0; t <= p; ++t) {        // only need layer1 h at timestep t==p
        // ---- stage x_t into LDS (16 rows x 384 halfs, 24 halfs/thread) ----
        {
            int r   = tid >> 4;
            int f0l = (tid & 15) * 24;
            const half8* src = (const half8*)(Xg + ((((long)p * NN + n0 + r) * LL + t) * F0 + f0l));
            half8* dst = (half8*)(&A0[r][f0l]);
            dst[0] = src[0]; dst[1] = src[1]; dst[2] = src[2];
        }
        __syncthreads();

        // ---- layer 0: gates = [x_t || h0] @ Wc0^T, K=512 (16 k-tiles) ----
        v8f acc[4];
        #pragma unroll
        for (int g = 0; g < 4; ++g) acc[g] = vzero;
        #pragma unroll 4
        for (int kt = 0; kt < 16; ++kt) {
            // t-dependent rotation of k-tile order: same sum, defeats LICM
            int ko = (((kt + 3 * t) & 15) << 5) + kb;
            half8 alo = *(const half8*)(&A0[arow][ko]);
            half8 ahi = *(const half8*)(&A0[arow][ko + 16]);
            v16h a = __builtin_shufflevector(alo, ahi, 0,1,2,3,4,5,6,7,8,9,10,11,12,13,14,15);
            #pragma unroll
            for (int g = 0; g < 4; ++g) {
                const _Float16* Bp = W0p + (long)(g * 128 + u) * K0 + ko;
                half8 blo = *(const half8*)Bp;
                half8 bhi = *(const half8*)(Bp + 16);
                v16h bv = __builtin_shufflevector(blo, bhi, 0,1,2,3,4,5,6,7,8,9,10,11,12,13,14,15);
                acc[g] = __builtin_amdgcn_wmma_f32_16x16x32_f16(
                    false, a, false, bv, (short)0, acc[g], false, false);
            }
        }
        __syncthreads();   // all GEMM reads of A0 complete before h writes

        float h0v[8];
        #pragma unroll
        for (int v = 0; v < 8; ++v) {
            float gi = sigf(acc[0][v] + bi0);
            float gf = sigf(acc[1][v] + bf0);
            float gg = tanhf(acc[2][v] + bg0);
            float go = sigf(acc[3][v] + bo0);
            c0s[v] = gf * c0s[v] + gi * gg;
            h0v[v] = go * tanhf(c0s[v]);
        }
        #pragma unroll
        for (int v = 0; v < 8; ++v) {
            int r = v + (hiH << 3);
            _Float16 hh = (_Float16)h0v[v];
            A0[r][F0 + u] = hh;   // h0 feeds next-step layer0
            A1[r][u]      = hh;   // h0_t feeds layer1 now
        }
        __syncthreads();

        // ---- layer 1: gates = [h0_t || h1] @ Wc1^T, K=256 (8 k-tiles) ----
        #pragma unroll
        for (int g = 0; g < 4; ++g) acc[g] = vzero;
        #pragma unroll 4
        for (int kt = 0; kt < 8; ++kt) {
            int ko = (((kt + 3 * t) & 7) << 5) + kb;
            half8 alo = *(const half8*)(&A1[arow][ko]);
            half8 ahi = *(const half8*)(&A1[arow][ko + 16]);
            v16h a = __builtin_shufflevector(alo, ahi, 0,1,2,3,4,5,6,7,8,9,10,11,12,13,14,15);
            #pragma unroll
            for (int g = 0; g < 4; ++g) {
                const _Float16* Bp = W1p + (long)(g * 128 + u) * K1 + ko;
                half8 blo = *(const half8*)Bp;
                half8 bhi = *(const half8*)(Bp + 16);
                v16h bv = __builtin_shufflevector(blo, bhi, 0,1,2,3,4,5,6,7,8,9,10,11,12,13,14,15);
                acc[g] = __builtin_amdgcn_wmma_f32_16x16x32_f16(
                    false, a, false, bv, (short)0, acc[g], false, false);
            }
        }
        __syncthreads();

        float h1v[8];
        #pragma unroll
        for (int v = 0; v < 8; ++v) {
            float gi = sigf(acc[0][v] + bi1);
            float gf = sigf(acc[1][v] + bf1);
            float gg = tanhf(acc[2][v] + bg1);
            float go = sigf(acc[3][v] + bo1);
            c1s[v] = gf * c1s[v] + gi * gg;
            h1v[v] = go * tanhf(c1s[v]);
        }
        #pragma unroll
        for (int v = 0; v < 8; ++v) {
            int r = v + (hiH << 3);
            A1[r][128 + u] = (_Float16)h1v[v];
            if (t == p) sel[((long)p * NN + n0 + r) * DD + u] = h1v[v];  // uniform branch
        }
        __syncthreads();
    }
}

// ---------------------------------------------------------------------------
// Kernel 4: MLP head + mean over paths + running-mean cumsum + embedding score
// ---------------------------------------------------------------------------
__global__ void score_kernel(const float* __restrict__ sel,
                             const float* __restrict__ W1, const float* __restrict__ b1v,
                             const float* __restrict__ W2, const float* __restrict__ b2v,
                             const int* __restrict__ cand, const int* __restrict__ user_index,
                             const float* __restrict__ news_emb, const float* __restrict__ user_emb,
                             float* __restrict__ out)
{
    __shared__ float psl[CC][PP];
    __shared__ float pm[CC];
    int b   = blockIdx.x;
    int tid = threadIdx.x;

    if (tid < CC * PP) {
        int c = tid / PP, p = tid % PP;
        int n = b * CC + c;
        const float* s = sel + ((long)p * NN + n) * DD;
        float o2 = b2v[0];
        for (int k = 0; k < 16; ++k) {
            float pre = b1v[k];
            const float* w = W1 + k * DD;
            for (int d = 0; d < DD; ++d) pre += s[d] * w[d];
            pre = pre > 0.f ? pre : 0.f;
            o2 += pre * W2[k];
        }
        psl[c][p] = sigf(o2);
    }
    __syncthreads();
    if (tid < CC) {
        float m = 0.f;
        for (int p = 0; p < PP; ++p) m += psl[tid][p];
        pm[tid] = m * (1.0f / PP);
    }
    __syncthreads();
    if (tid < CC) {
        int c = tid;
        float cs = 0.f;
        for (int c2 = 0; c2 <= c; ++c2) cs += pm[c2];
        float ts = cs / (float)(c + 1);
        int ni = cand[b * CC + c];
        int ui = user_index[b];
        const float* ne = news_emb + (long)ni * DD;
        const float* ue = user_emb + (long)ui * DD;
        float dot = 0.f;
        for (int d = 0; d < DD; ++d) dot += ne[d] * ue[d];
        out[b * CC + c] = sigf(dot) + ts;
    }
}

// ---------------------------------------------------------------------------
extern "C" void kernel_launch(void* const* d_in, const int* in_sizes, int n_in,
                              void* d_out, int out_size, void* d_ws, size_t ws_size,
                              hipStream_t stream)
{
    (void)in_sizes; (void)n_in; (void)out_size; (void)ws_size;

    const int*   cand         = (const int*)d_in[0];
    const int*   uidx         = (const int*)d_in[1];
    const int*   paths        = (const int*)d_in[2];
    const int*   types        = (const int*)d_in[3];
    const int*   rels         = (const int*)d_in[4];
    const float* entity_emb   = (const float*)d_in[5];
    const float* relation_emb = (const float*)d_in[6];
    const float* user_emb     = (const float*)d_in[7];
    const float* news_emb     = (const float*)d_in[8];
    const float* type_emb     = (const float*)d_in[9];
    const float* Wih0         = (const float*)d_in[10];
    const float* Whh0         = (const float*)d_in[11];
    const float* bih0         = (const float*)d_in[12];
    const float* bhh0         = (const float*)d_in[13];
    const float* Wih1         = (const float*)d_in[14];
    const float* Whh1         = (const float*)d_in[15];
    const float* bih1         = (const float*)d_in[16];
    const float* bhh1         = (const float*)d_in[17];
    const float* W1           = (const float*)d_in[18];
    const float* b1           = (const float*)d_in[19];
    const float* W2           = (const float*)d_in[20];
    const float* b2           = (const float*)d_in[21];

    char* ws = (char*)d_ws;
    _Float16* Xg  = (_Float16*)ws;  ws += (size_t)PP * NN * LL * F0 * 2;   // 12.29 MB
    _Float16* Wc0 = (_Float16*)ws;  ws += (size_t)PP * G4 * K0 * 2;        //  2.62 MB
    _Float16* Wc1 = (_Float16*)ws;  ws += (size_t)PP * G4 * K1 * 2;        //  1.31 MB
    float*    b0c = (float*)ws;     ws += (size_t)PP * G4 * 4;
    float*    b1c = (float*)ws;     ws += (size_t)PP * G4 * 4;
    float*    selb = (float*)ws;    ws += (size_t)PP * NN * DD * 4;        //  1.64 MB

    build_x_kernel<<<BB * CC * PP * LL, 128, 0, stream>>>(
        uidx, paths, types, rels, entity_emb, relation_emb, user_emb, news_emb, type_emb, Xg);

    build_w_kernel<<<PP * G4, 256, 0, stream>>>(
        Wih0, Whh0, bih0, bhh0, Wih1, Whh1, bih1, bhh1, Wc0, Wc1, b0c, b1c);

    dim3 grid(NN / 16, PP);
    lstm_kernel<<<grid, 256, 0, stream>>>(Xg, Wc0, Wc1, b0c, b1c, selb);

    score_kernel<<<BB, 128, 0, stream>>>(
        selb, W1, b1, W2, b2, cand, uidx, news_emb, user_emb, (float*)d_out);
}
